// EncoderLayer_32770600468860
// MI455X (gfx1250) — matrix operand order repaired and checked by reading, not compiled, not run on hardware
//
#include <hip/hip_runtime.h>
#include <hip/hip_bf16.h>

#define BB 8
#define SS 1024
#define DD 1024
#define HH 16
#define DKK 64

typedef __bf16 bf16_t;
typedef __bf16 v16bf __attribute__((ext_vector_type(16)));
typedef __bf16 v8bf  __attribute__((ext_vector_type(8)));
typedef float  v8f   __attribute__((ext_vector_type(8)));
typedef int    v4i   __attribute__((ext_vector_type(4)));
typedef unsigned int u32x4 __attribute__((ext_vector_type(4)));
typedef int i32x4 __attribute__((ext_vector_type(4)));
typedef int i32x8 __attribute__((ext_vector_type(8)));

#define ZERO8F {0.f,0.f,0.f,0.f,0.f,0.f,0.f,0.f}

#if __has_builtin(__builtin_amdgcn_global_load_async_to_lds_b128)
#define HAS_ASYNC 1
#else
#define HAS_ASYNC 0
#endif
#if __has_builtin(__builtin_amdgcn_tensor_load_to_lds)
#define HAS_TDM 1
#else
#define HAS_TDM 0
#endif

#define AS3V(p)    ((__attribute__((address_space(3))) void*)(p))
#define ASG_V4I(p) ((__attribute__((address_space(1))) v4i*)(p))
#define ASL_V4I(p) ((__attribute__((address_space(3))) v4i*)(p))

__device__ __forceinline__ v8f wmma_bf16(v16bf a, v16bf b, v8f c) {
  return __builtin_amdgcn_wmma_f32_16x16x32_bf16(false, a, false, b, (short)0, c,
                                                 false, false);
}

__device__ __forceinline__ v16bf mkfrag(v8bf lo, v8bf hi) {
  v16bf r;
#pragma unroll
  for (int i = 0; i < 8; ++i) { r[i] = lo[i]; r[i + 8] = hi[i]; }
  return r;
}

__device__ __forceinline__ v8bf ldbf8(const bf16_t* p) { return *(const v8bf*)p; }

// 16B global -> LDS copy, async (direct-to-LDS DMA) when available.
__device__ __forceinline__ void async_copy16(const bf16_t* g, bf16_t* l) {
#if HAS_ASYNC
  __builtin_amdgcn_global_load_async_to_lds_b128(ASG_V4I(g), ASL_V4I(l), 0, 0);
#else
  *(v8bf*)l = ldbf8(g);
#endif
}

__device__ __forceinline__ void async_join() {
#if HAS_ASYNC
#if __has_builtin(__builtin_amdgcn_s_wait_asynccnt)
  __builtin_amdgcn_s_wait_asynccnt(0);
#else
  asm volatile("s_wait_asynccnt 0x0" ::: "memory");
#endif
#endif
}

#if HAS_TDM
// 2D Tensor-DMA: tile_x (elems along contiguous dim) x tile_y rows of 2-byte
// elements, source row stride = stride_x elems, with LDS row padding of
// pad_amount DWORDs after each pad_interval-coded group (codes per D# spec).
__device__ __forceinline__ void tdm_load_tile_2d(unsigned lds_off,
                                                 const bf16_t* gaddr,
                                                 unsigned tile_x, unsigned tile_y,
                                                 unsigned tensor_x, unsigned tensor_y,
                                                 unsigned stride_x,
                                                 unsigned pad_interval_code,
                                                 unsigned pad_amount_code,
                                                 bool pad_en) {
  unsigned long long ga = (unsigned long long)gaddr;
  u32x4 g0;
  g0[0] = 1u;                                              // count=1, load D#
  g0[1] = lds_off;                                         // lds_addr
  g0[2] = (unsigned)(ga & 0xffffffffu);                    // global_addr[31:0]
  g0[3] = (unsigned)((ga >> 32) & 0x01ffffffu) | (2u << 30); // addr[56:32]|type=2
  i32x8 g1;
  unsigned w0 = (1u << 16);                                // data_size=1 -> 2B
  if (pad_en) w0 |= (1u << 20) | (pad_interval_code << 22) | (pad_amount_code << 25);
  g1[0] = (int)w0;
  g1[1] = (int)((tensor_x & 0xffffu) << 16);               // tensor_dim0[15:0]
  g1[2] = (int)(((tensor_x >> 16) & 0xffffu) | ((tensor_y & 0xffffu) << 16));
  g1[3] = (int)(((tensor_y >> 16) & 0xffffu) | ((tile_x & 0xffffu) << 16));
  g1[4] = (int)(tile_y & 0xffffu);                         // tile_dim1, tile_dim2=0
  g1[5] = (int)stride_x;                                   // dim0_stride[31:0]
  g1[6] = 0;                                               // stride hi, dim1_stride lo
  g1[7] = 0;
  i32x4 z4 = {0, 0, 0, 0};
#if __clang_major__ >= 23
  i32x8 z8 = {0, 0, 0, 0, 0, 0, 0, 0};
  __builtin_amdgcn_tensor_load_to_lds(g0, g1, z4, z4, z8, 0);
#else
  __builtin_amdgcn_tensor_load_to_lds(g0, g1, z4, z4, 0);
#endif
}

__device__ __forceinline__ void tdm_wait() {
#if __has_builtin(__builtin_amdgcn_s_wait_tensorcnt)
  __builtin_amdgcn_s_wait_tensorcnt(0);
#else
  asm volatile("s_wait_tensorcnt 0x0" ::: "memory");
#endif
}
#endif  // HAS_TDM

// ---------------------------------------------------------------------------
// LayerNorm (torch semantics), optional residual, optional f32/bf16 outputs.
// ---------------------------------------------------------------------------
__global__ __launch_bounds__(256) void k_norm2(const float* __restrict__ a,
                                               const float* __restrict__ b,
                                               const float* __restrict__ alpha,
                                               const float* __restrict__ beta,
                                               float* __restrict__ outf,
                                               bf16_t* __restrict__ outb,
                                               float oscale) {
  const size_t base = (size_t)blockIdx.x * DD;
  float v[4];
  float s = 0.f, s2 = 0.f;
#pragma unroll
  for (int i = 0; i < 4; ++i) {
    int idx = threadIdx.x + i * 256;
    float t = a[base + idx];
    if (b) t += b[base + idx];
    v[i] = t; s += t; s2 += t * t;
  }
  __shared__ float sm[256], sq[256];
  sm[threadIdx.x] = s; sq[threadIdx.x] = s2;
  __syncthreads();
  for (int off = 128; off > 0; off >>= 1) {
    if (threadIdx.x < (unsigned)off) {
      sm[threadIdx.x] += sm[threadIdx.x + off];
      sq[threadIdx.x] += sq[threadIdx.x + off];
    }
    __syncthreads();
  }
  float mu  = sm[0] * (1.f / DD);
  float var = (sq[0] - (float)DD * mu * mu) * (1.f / (DD - 1));
  var = var < 0.f ? 0.f : var;
  float inv = 1.f / (sqrtf(var) + 1e-5f);
#pragma unroll
  for (int i = 0; i < 4; ++i) {
    int idx = threadIdx.x + i * 256;
    float y = (alpha[idx] * (v[i] - mu) * inv + beta[idx]) * oscale;
    if (outf) outf[base + idx] = y;
    if (outb) outb[base + idx] = (bf16_t)y;
  }
}

__global__ void k_cast(const float* __restrict__ w, bf16_t* __restrict__ o, int n) {
  int i = blockIdx.x * blockDim.x + threadIdx.x;
  if (i < n) o[i] = (bf16_t)w[i];
}

// ---------------------------------------------------------------------------
// GEMM: C[M,N] = A[M,K](bf16) * Bw[K,N](bf16) + bias. 64x64 block, 4 waves.
// A tile staged via TDM (per-wave 16-row strip descriptors) or async-to-LDS;
// B tile transposed in-regs so fragment gathers are contiguous ds_load_b128.
// ---------------------------------------------------------------------------
__global__ __launch_bounds__(128) void k_gemm(const bf16_t* __restrict__ A,
                                              const bf16_t* __restrict__ Bw,
                                              const float* __restrict__ bias,
                                              float* __restrict__ Cf,
                                              bf16_t* __restrict__ Cb,
                                              int M, int N, int K, int store_bf16) {
  __shared__ __attribute__((aligned(16))) bf16_t sA[64][40];   // [m][k], pad 8
  __shared__ __attribute__((aligned(16))) bf16_t sBt[64][40];  // [n][k], pad 8
  const int tid = threadIdx.x;
  const int lane = tid & 31, wave = tid >> 5;
  const int m0 = blockIdx.y * 64, n0 = blockIdx.x * 64;
  const int ml = lane & 15, hi = lane >> 4;
  const int kg0 = hi ? 8 : 0, kg1 = hi ? 24 : 16;  // A-frag K groups
  const int kb  = hi ? 16 : 0;                     // unused; B-frag also uses kg0/kg1

  v8f acc[4];
#pragma unroll
  for (int t = 0; t < 4; ++t) { v8f z = ZERO8F; acc[t] = z; }

#if HAS_TDM
  const unsigned sA_lds = (unsigned)(unsigned long long)AS3V(&sA[0][0]);
#endif

  for (int k0 = 0; k0 < K; k0 += 32) {
#if HAS_TDM
    // Per-wave TDM: 32(k) x 16(m) tile strip; pad 4 DWORDs after every 16
    // DWORDs so LDS rows land on the 40-element padded pitch.
    tdm_load_tile_2d(sA_lds + (unsigned)(wave * 16) * 40u * 2u,
                     A + (size_t)(m0 + wave * 16) * K + k0,
                     /*tile_x=*/32, /*tile_y=*/16,
                     /*tensor_x=*/(unsigned)K, /*tensor_y=*/(unsigned)M,
                     /*stride_x=*/(unsigned)K,
                     /*pad_interval(16dw)=*/3, /*pad_amount(4dw)=*/3, true);
#else
#pragma unroll
    for (int it = 0; it < 2; ++it) {   // A tile: 64x32, 16B chunks
      int c = tid + it * 128;
      int r = c >> 2, cc = (c & 3) * 8;
      const bf16_t* g = A + (size_t)(m0 + r) * K + k0 + cc;
      async_copy16(g, &sA[r][cc]);
      if (k0 + 32 < K) __builtin_prefetch(g + 32, 0, 3);
    }
#endif
#pragma unroll
    for (int it = 0; it < 2; ++it) {   // B tile: 32x64, transpose into sBt
      int c = tid + it * 128;
      int kk = c >> 3, nn = (c & 7) * 8;
      v8bf t = ldbf8(Bw + (size_t)(k0 + kk) * N + n0 + nn);
#pragma unroll
      for (int j = 0; j < 8; ++j) sBt[nn + j][kk] = t[j];
    }
#if HAS_TDM
    tdm_wait();
#else
    async_join();
#endif
    __syncthreads();

    int mrow = wave * 16 + ml;
    v16bf a = mkfrag(ldbf8(&sA[mrow][kg0]), ldbf8(&sA[mrow][kg1]));
#pragma unroll
    for (int nt = 0; nt < 4; ++nt) {
      int n = nt * 16 + ml;
      v16bf bf = mkfrag(ldbf8(&sBt[n][kg0]), ldbf8(&sBt[n][kg1]));
      acc[nt] = wmma_bf16(a, bf, acc[nt]);
    }
    __syncthreads();
  }

#pragma unroll
  for (int nt = 0; nt < 4; ++nt) {
    int n = n0 + nt * 16 + ml;
    float bvv = bias[n];
#pragma unroll
    for (int r = 0; r < 8; ++r) {
      int m = m0 + wave * 16 + (hi ? 8 + r : r);
      float val = acc[nt][r] + bvv;
      if (store_bf16) Cb[(size_t)m * N + n] = (bf16_t)val;
      else            Cf[(size_t)m * N + n] = val;
    }
  }
}

// ---------------------------------------------------------------------------
// Flash attention: block = (qtile=64, head, batch), 4 waves, wave owns 16 rows.
// K tile staged with async-to-LDS (overlaps with in-reg V transpose).
// ---------------------------------------------------------------------------
__global__ __launch_bounds__(128) void k_attn(const bf16_t* __restrict__ Qm,
                                              const bf16_t* __restrict__ Km,
                                              const bf16_t* __restrict__ Vm,
                                              const int* __restrict__ mask,
                                              bf16_t* __restrict__ Om) {
  __shared__ __attribute__((aligned(16))) bf16_t sK[32][72];      // [key][dk]
  __shared__ __attribute__((aligned(16))) bf16_t sVt[64][40];     // [dk][key]
  __shared__ __attribute__((aligned(16))) bf16_t sP[4][16][40];   // per-wave P
  const int tid = threadIdx.x, lane = tid & 31, wave = tid >> 5;
  const int bz = blockIdx.z, h = blockIdx.y, q0 = blockIdx.x * 64;
  const int ml = lane & 15, hi = lane >> 4;
  const int kg0 = hi ? 8 : 0, kg1 = hi ? 24 : 16;
  const int kb  = hi ? 16 : 0;
  const size_t hoff = (size_t)h * DKK;

  const bf16_t* qrow = Qm + ((size_t)(bz * SS + q0 + wave * 16 + ml)) * DD + hoff;
  v16bf aq0 = mkfrag(ldbf8(qrow + kg0),      ldbf8(qrow + kg1));
  v16bf aq1 = mkfrag(ldbf8(qrow + 32 + kg0), ldbf8(qrow + 32 + kg1));

  v8f oacc[4];
#pragma unroll
  for (int t = 0; t < 4; ++t) { v8f z = ZERO8F; oacc[t] = z; }
  float rmax[8], rsum[8];
#pragma unroll
  for (int r = 0; r < 8; ++r) { rmax[r] = -1e30f; rsum[r] = 0.f; }

  for (int kt = 0; kt < SS / 32; ++kt) {
    __syncthreads();
#pragma unroll
    for (int it = 0; it < 2; ++it) {  // K tile 32x64 row-major (async DMA)
      int c = tid + it * 128;
      int r = c >> 3, cc = (c & 7) * 8;
      async_copy16(Km + ((size_t)(bz * SS + kt * 32 + r)) * DD + hoff + cc,
                   &sK[r][cc]);
    }
#pragma unroll
    for (int it = 0; it < 2; ++it) {  // V tile transposed -> sVt[dk][key]
      int c = tid + it * 128;
      int key = c >> 3, dd = (c & 7) * 8;
      v8bf t = ldbf8(Vm + ((size_t)(bz * SS + kt * 32 + key)) * DD + hoff + dd);
#pragma unroll
      for (int j = 0; j < 8; ++j) sVt[dd + j][key] = t[j];
    }
    async_join();
    __syncthreads();

    // scores: two 16x16 subtiles, each = QK^T over DK=64 (2 WMMAs)
    float sv[2][8];
#pragma unroll
    for (int sub = 0; sub < 2; ++sub) {
      int keyn = sub * 16 + ml;
      v16bf b0 = mkfrag(ldbf8(&sK[keyn][kg0]),      ldbf8(&sK[keyn][kg1]));
      v16bf b1 = mkfrag(ldbf8(&sK[keyn][32 + kg0]), ldbf8(&sK[keyn][32 + kg1]));
      v8f sc = ZERO8F;
      sc = wmma_bf16(aq0, b0, sc);
      sc = wmma_bf16(aq1, b1, sc);
      int mk = mask[bz * SS + kt * 32 + keyn];
#pragma unroll
      for (int r = 0; r < 8; ++r)
        sv[sub][r] = mk ? sc[r] * 0.125f : -1e9f;
    }

    // online softmax per row (rows live in 16-lane halves per C layout)
#pragma unroll
    for (int r = 0; r < 8; ++r) {
      float tm = fmaxf(sv[0][r], sv[1][r]);
#pragma unroll
      for (int mo = 8; mo > 0; mo >>= 1) tm = fmaxf(tm, __shfl_xor(tm, mo, 32));
      float nm = fmaxf(rmax[r], tm);
      float corr = __expf(rmax[r] - nm);
      rmax[r] = nm;
      float p0 = __expf(sv[0][r] - nm);
      float p1 = __expf(sv[1][r] - nm);
      float ts = p0 + p1;
#pragma unroll
      for (int mo = 8; mo > 0; mo >>= 1) ts += __shfl_xor(ts, mo, 32);
      rsum[r] = rsum[r] * corr + ts;
#pragma unroll
      for (int nt = 0; nt < 4; ++nt) oacc[nt][r] *= corr;
      int m = hi ? 8 + r : r;
      sP[wave][m][ml]      = (bf16_t)p0;
      sP[wave][m][16 + ml] = (bf16_t)p1;
    }

    // O += P(16x32) * V(32x64)
    v16bf ap = mkfrag(ldbf8(&sP[wave][ml][kg0]), ldbf8(&sP[wave][ml][kg1]));
#pragma unroll
    for (int nt = 0; nt < 4; ++nt) {
      int n = nt * 16 + ml;
      v16bf bv = mkfrag(ldbf8(&sVt[n][kg0]), ldbf8(&sVt[n][kg1]));
      oacc[nt] = wmma_bf16(ap, bv, oacc[nt]);
    }
  }

#pragma unroll
  for (int nt = 0; nt < 4; ++nt) {
    int dk = nt * 16 + ml;
#pragma unroll
    for (int r = 0; r < 8; ++r) {
      int m = hi ? 8 + r : r;
      float v = oacc[nt][r] / rsum[r];
      Om[((size_t)(bz * SS + q0 + wave * 16 + m)) * DD + hoff + dk] = (bf16_t)v;
    }
  }
}

// ---------------------------------------------------------------------------
extern "C" void kernel_launch(void* const* d_in, const int* in_sizes, int n_in,
                              void* d_out, int out_size, void* d_ws, size_t ws_size,
                              hipStream_t stream) {
  (void)in_sizes; (void)n_in; (void)out_size; (void)ws_size;
  const float* x      = (const float*)d_in[0];
  const int*   mask   = (const int*)d_in[1];
  const float* alpha1 = (const float*)d_in[2];
  const float* bias1  = (const float*)d_in[3];
  const float* alpha2 = (const float*)d_in[4];
  const float* bias2  = (const float*)d_in[5];
  const float* Wq = (const float*)d_in[6];   const float* bq = (const float*)d_in[7];
  const float* Wk = (const float*)d_in[8];   const float* bk = (const float*)d_in[9];
  const float* Wv = (const float*)d_in[10];  const float* bv = (const float*)d_in[11];
  const float* Wo = (const float*)d_in[12];  const float* bo = (const float*)d_in[13];
  float* out = (float*)d_out;

  const size_t NTOK = (size_t)BB * SS;
  const size_t szF = NTOK * DD * sizeof(float);
  const size_t szH = NTOK * DD * sizeof(bf16_t);
  const size_t szW = (size_t)DD * DD * sizeof(bf16_t);

  char* p = (char*)d_ws;
  float*  xn  = (float*)p;   p += szF;
  bf16_t* xnb = (bf16_t*)p;  p += szH;
  bf16_t* wqb = (bf16_t*)p;  p += szW;
  bf16_t* wkb = (bf16_t*)p;  p += szW;
  bf16_t* wvb = (bf16_t*)p;  p += szW;
  bf16_t* wob = (bf16_t*)p;  p += szW;
  bf16_t* qb  = (bf16_t*)p;  p += szH;
  bf16_t* kbf = (bf16_t*)p;  p += szH;
  bf16_t* vbf = (bf16_t*)p;  p += szH;
  bf16_t* obf = (bf16_t*)p;  p += szH;
  float*  att = (float*)p;   p += szF;

  k_norm2<<<(int)NTOK, 256, 0, stream>>>(x, nullptr, alpha1, bias1, xn, xnb, 1.f);

  const int NW = DD * DD;
  k_cast<<<(NW + 255) / 256, 256, 0, stream>>>(Wq, wqb, NW);
  k_cast<<<(NW + 255) / 256, 256, 0, stream>>>(Wk, wkb, NW);
  k_cast<<<(NW + 255) / 256, 256, 0, stream>>>(Wv, wvb, NW);
  k_cast<<<(NW + 255) / 256, 256, 0, stream>>>(Wo, wob, NW);

  dim3 gg(DD / 64, (int)(NTOK / 64));
  k_gemm<<<gg, 128, 0, stream>>>(xnb, wqb, bq, nullptr, qb, (int)NTOK, DD, DD, 1);
  k_gemm<<<gg, 128, 0, stream>>>(xnb, wkb, bk, nullptr, kbf, (int)NTOK, DD, DD, 1);
  k_gemm<<<gg, 128, 0, stream>>>(xnb, wvb, bv, nullptr, vbf, (int)NTOK, DD, DD, 1);

  dim3 ga(SS / 64, HH, BB);
  k_attn<<<ga, 128, 0, stream>>>(qb, kbf, vbf, mask, obf);

  k_gemm<<<gg, 128, 0, stream>>>(obf, wob, bo, att, nullptr, (int)NTOK, DD, DD, 0);

  k_norm2<<<(int)NTOK, 256, 0, stream>>>(xn, att, alpha2, bias2, out, nullptr, 2.f);
}
